// FactoredCausalSelfAttention_39582418600153
// MI455X (gfx1250) — matrix operand order repaired and checked by reading, not compiled, not run on hardware
//
#include <hip/hip_runtime.h>
#include <hip/hip_bf16.h>
#include <stdint.h>

typedef __attribute__((ext_vector_type(16))) __bf16 v16bf;
typedef __attribute__((ext_vector_type(8)))  __bf16 v8bf;
typedef __attribute__((ext_vector_type(8)))  float  v8f;

#define BB 4
#define TT 2048
#define CC 1024
#define HH 16
#define DD 64

// ---------------- CDNA5 async global->LDS copy (tracked by ASYNCcnt) ----------------
static __device__ inline void async_b128(uint32_t lds_byte_off, const void* gaddr) {
  asm volatile("global_load_async_to_lds_b128 %0, %1, off"
               :: "v"(lds_byte_off), "v"(gaddr) : "memory");
}
static __device__ inline void wait_async() {
  asm volatile("s_wait_asynccnt 0x0" ::: "memory");
}
static __device__ inline void wait_ds() {
  asm volatile("s_wait_dscnt 0x0" ::: "memory");
}
static __device__ inline uint32_t lds_off(const void* p) {
  return (uint32_t)(size_t)p;  // low 32 bits of flat LDS address = LDS byte offset
}

static __device__ inline v8f v8f_zero() {
  v8f z;
#pragma unroll
  for (int i = 0; i < 8; ++i) z[i] = 0.0f;
  return z;
}

static __device__ inline v8f wmma_bf16f32(v16bf a, v16bf b, v8f c) {
  return __builtin_amdgcn_wmma_f32_16x16x32_bf16(
      false, a, false, b, (short)0, c, false, false);
}

// ---- Fragment loaders (wave32, contiguous k = 16h + s order shared by A and B), all-bf16 sources ----
// A 16x32: lanes 0-15 row=lane K=0..15; lanes 16-31 row=lane-16 K=16..31
static __device__ inline v16bf a_frag_bf16(const __bf16* base, int ld, int row0, int k0, int lane) {
  const __bf16* p = base + (size_t)(row0 + (lane & 15)) * ld + k0 + ((lane >> 4) << 4);
  v16bf a;
#pragma unroll
  for (int i = 0; i < 8; ++i) a[i] = p[i];
#pragma unroll
  for (int i = 0; i < 8; ++i) a[8 + i] = p[8 + i];
  return a;
}
// B 32x16 from row-major [N, ld] (k contiguous): lanes 0-15 col=lane K=k0..+15; lanes 16-31 K=k0+16..+31
static __device__ inline v16bf b_frag_bf16(const __bf16* M, int ld, int n0, int k0, int lane) {
  const __bf16* p = M + (size_t)(n0 + (lane & 15)) * ld + k0 + ((lane >> 4) << 4);
  v16bf b;
#pragma unroll
  for (int i = 0; i < 16; ++i) b[i] = p[i];
  return b;
}

// ================= Kernel 0: one-time f32 -> bf16 conversion (bandwidth-bound) ==========
__global__ __launch_bounds__(256) void cvt_bf16_kernel(const float* __restrict__ src,
                                                       __bf16* __restrict__ dst) {
  const size_t i = ((size_t)blockIdx.x * 256 + threadIdx.x) * 8;
  const float4* s = (const float4*)(src + i);
  const float4 v0 = s[0];
  const float4 v1 = s[1];
  v8bf o;
  o[0] = (__bf16)v0.x; o[1] = (__bf16)v0.y; o[2] = (__bf16)v0.z; o[3] = (__bf16)v0.w;
  o[4] = (__bf16)v1.x; o[5] = (__bf16)v1.y; o[6] = (__bf16)v1.z; o[7] = (__bf16)v1.w;
  *(v8bf*)(dst + i) = o;
}

// ================= Kernel 1: QKV projection, head scatter, V gating =================
// WG tile 128Mx128N, 8 waves as 4(M)x2(N), wave tile 32x64, K-tile 64 (16 WMMA / barrier).
// Double-buffered async bf16 LDS staging. Q*0.125 -> [B,H,T,D]; K -> [B,H,T,D];
// effV = xt*v -> d-major [B,H,D,T].
__global__ __launch_bounds__(256) void qkv_gemm_kernel(
    const __bf16* __restrict__ xbf, const float* __restrict__ xt,
    const __bf16* __restrict__ Wab, const float* __restrict__ battn,
    __bf16* __restrict__ Qh, __bf16* __restrict__ Kh, __bf16* __restrict__ Vh)
{
  __shared__ __bf16 As[2][128][64];   // 16KB per buffer
  __shared__ __bf16 Bs[2][128][64];

  const int tid  = threadIdx.x;
  const int lane = tid & 31;
  const int wave = tid >> 5;
  const int n0 = blockIdx.x * 128;
  const int m0 = blockIdx.y * 128;
  const int wm = (wave >> 1) * 32;
  const int wn = (wave & 1) * 64;

  const int crow = tid >> 1;          // 0..127
  const int cseg = (tid & 1) * 32;    // bf16 elements (32 bf16 = 64B per thread)

  auto issue_tile = [&](int k0, int buf) {
    const __bf16* gA = xbf + (size_t)(m0 + crow) * CC + k0 + cseg;
    uint32_t la = lds_off(&As[buf][crow][cseg]);
    async_b128(la,      gA);
    async_b128(la + 16, gA + 8);
    async_b128(la + 32, gA + 16);
    async_b128(la + 48, gA + 24);
    const __bf16* gB = Wab + (size_t)(n0 + crow) * CC + k0 + cseg;
    uint32_t lb = lds_off(&Bs[buf][crow][cseg]);
    async_b128(lb,      gB);
    async_b128(lb + 16, gB + 8);
    async_b128(lb + 32, gB + 16);
    async_b128(lb + 48, gB + 24);
  };

  v8f acc[2][4];
#pragma unroll
  for (int a = 0; a < 2; ++a)
#pragma unroll
    for (int j = 0; j < 4; ++j) acc[a][j] = v8f_zero();

  int buf = 0;
  issue_tile(0, buf);
  for (int kt = 0; kt < CC / 64; ++kt) {
    wait_async();
    __syncthreads();
    if (kt + 1 < CC / 64) issue_tile((kt + 1) * 64, buf ^ 1);
    const __bf16* Al = &As[buf][0][0];
    const __bf16* Bl = &Bs[buf][0][0];
#pragma unroll
    for (int kk = 0; kk < 2; ++kk) {
      v16bf a0 = a_frag_bf16(Al, 64, wm,      kk * 32, lane);
      v16bf a1 = a_frag_bf16(Al, 64, wm + 16, kk * 32, lane);
#pragma unroll
      for (int j = 0; j < 4; ++j) {
        v16bf b = b_frag_bf16(Bl, 64, wn + 16 * j, kk * 32, lane);
        acc[0][j] = wmma_bf16f32(a0, b, acc[0][j]);
        acc[1][j] = wmma_bf16f32(a1, b, acc[1][j]);
      }
    }
    buf ^= 1;
  }

  const int rbase = (lane >> 4) << 3;
  const int nl = lane & 15;
#pragma unroll
  for (int aT = 0; aT < 2; ++aT) {
#pragma unroll
    for (int j = 0; j < 4; ++j) {
      const int ncol = n0 + wn + 16 * j + nl;   // 0..3071
      const float bv = battn[ncol];
      const int which = ncol >> 10;             // 0=Q 1=K 2=V
      const int cc2 = ncol & (CC - 1);
      const int h = cc2 >> 6;
      const int d = cc2 & 63;
#pragma unroll
      for (int r = 0; r < 8; ++r) {
        const int row = m0 + wm + 16 * aT + rbase + r;  // b*T + t
        const int bidx = row >> 11;
        const int t = row & (TT - 1);
        const float val = acc[aT][j][r] + bv;
        if (which == 0) {
          Qh[(((size_t)bidx * HH + h) * TT + t) * DD + d] = (__bf16)(val * 0.125f);
        } else if (which == 1) {
          Kh[(((size_t)bidx * HH + h) * TT + t) * DD + d] = (__bf16)val;
        } else {
          const float g = xt[(size_t)row * CC + cc2];
          Vh[(((size_t)bidx * HH + h) * DD + d) * TT + t] = (__bf16)(val * g);
        }
      }
    }
  }
}

// ================= Kernel 2: causal flash attention =================
__global__ __launch_bounds__(256) void flash_attn_kernel(
    const __bf16* __restrict__ Qh, const __bf16* __restrict__ Kh,
    const __bf16* __restrict__ Vh, __bf16* __restrict__ Y)
{
  __shared__ __bf16 Ks[2][64][64];   // [key][d]
  __shared__ __bf16 Vs[2][64][64];   // [d][key]
  __shared__ __bf16 Ps[8][16 * 64];  // per-wave P tile

  const int tid  = threadIdx.x;
  const int lane = tid & 31;
  const int wave = tid >> 5;
  const int bh = blockIdx.y;
  const int b = bh >> 4;
  const int h = bh & 15;
  const __bf16* Qb = Qh + (size_t)bh * TT * DD;
  const __bf16* Kb = Kh + (size_t)bh * TT * DD;
  const __bf16* Vb = Vh + (size_t)bh * DD * TT;   // d-major [D][T]

  const int qblk = blockIdx.x * 128;
  const int q0 = qblk + wave * 16;
  const int rbase = (lane >> 4) << 3;
  const int nl = lane & 15;
  const float NEG = -3.0e38f;

  const int crow = tid >> 2;          // 0..63
  const int cseg = (tid & 3) * 16;    // 16 bf16 = 32B per thread

  auto issue_kv = [&](int s0, int buf) {
    const __bf16* gK = Kb + (size_t)(s0 + crow) * DD + cseg;
    uint32_t lk = lds_off(&Ks[buf][crow][cseg]);
    async_b128(lk,      gK);
    async_b128(lk + 16, gK + 8);
    const __bf16* gV = Vb + (size_t)crow * TT + s0 + cseg;  // row = d, cols = keys
    uint32_t lv = lds_off(&Vs[buf][crow][cseg]);
    async_b128(lv,      gV);
    async_b128(lv + 16, gV + 8);
  };

  const v16bf aq0 = a_frag_bf16(Qb, DD, q0, 0, lane);
  const v16bf aq1 = a_frag_bf16(Qb, DD, q0, 32, lane);

  v8f acc[4];
  float mrow[8], lrow[8];
#pragma unroll
  for (int j = 0; j < 4; ++j) acc[j] = v8f_zero();
#pragma unroll
  for (int r = 0; r < 8; ++r) { mrow[r] = NEG; lrow[r] = 0.0f; }
  __bf16* pb = &Ps[wave][0];

  const int nblocks = (qblk + 128) / 64;  // uniform across the workgroup
  int buf = 0;
  issue_kv(0, buf);
  for (int it = 0; it < nblocks; ++it) {
    const int s0 = it * 64;
    wait_async();
    __syncthreads();
    if (it + 1 < nblocks) issue_kv(s0 + 64, buf ^ 1);

    const __bf16* Kl = &Ks[buf][0][0];
    const __bf16* Vl = &Vs[buf][0][0];

    // ---- S = (Q/sqrt(D)) @ K^T : 16 x 64 ----
    v8f sc[4];
#pragma unroll
    for (int j = 0; j < 4; ++j) sc[j] = v8f_zero();
#pragma unroll
    for (int j = 0; j < 4; ++j) {
      sc[j] = wmma_bf16f32(aq0, b_frag_bf16(Kl, 64, 16 * j, 0, lane), sc[j]);
      sc[j] = wmma_bf16f32(aq1, b_frag_bf16(Kl, 64, 16 * j, 32, lane), sc[j]);
    }

    // ---- causal mask + online softmax (D-layout; fully-masked blocks contribute 0) ----
#pragma unroll
    for (int r = 0; r < 8; ++r) {
      const int qg = q0 + rbase + r;
      float mx = NEG;
#pragma unroll
      for (int j = 0; j < 4; ++j) {
        const int sg = s0 + 16 * j + nl;
        const float v = (sg <= qg) ? sc[j][r] : NEG;
        sc[j][r] = v;
        mx = fmaxf(mx, v);
      }
      mx = fmaxf(mx, __shfl_xor(mx, 1, 32));
      mx = fmaxf(mx, __shfl_xor(mx, 2, 32));
      mx = fmaxf(mx, __shfl_xor(mx, 4, 32));
      mx = fmaxf(mx, __shfl_xor(mx, 8, 32));
      const float mnew = fmaxf(mrow[r], mx);
      const float alpha = __expf(mrow[r] - mnew);
      mrow[r] = mnew;
      float rs = 0.0f;
#pragma unroll
      for (int j = 0; j < 4; ++j) {
        const float p = __expf(sc[j][r] - mnew);
        sc[j][r] = p;
        rs += p;
      }
      rs += __shfl_xor(rs, 1, 32);
      rs += __shfl_xor(rs, 2, 32);
      rs += __shfl_xor(rs, 4, 32);
      rs += __shfl_xor(rs, 8, 32);
      lrow[r] = lrow[r] * alpha + rs;
#pragma unroll
      for (int jd = 0; jd < 4; ++jd) acc[jd][r] *= alpha;
    }

    // ---- P: D-layout -> A-layout via per-wave LDS tile ----
#pragma unroll
    for (int j = 0; j < 4; ++j)
#pragma unroll
      for (int r = 0; r < 8; ++r)
        pb[(rbase + r) * 64 + 16 * j + nl] = (__bf16)sc[j][r];
    wait_ds();

    // ---- acc += P @ effV ----
#pragma unroll
    for (int kk = 0; kk < 2; ++kk) {
      v16bf ap = a_frag_bf16(pb, 64, 0, kk * 32, lane);
#pragma unroll
      for (int jd = 0; jd < 4; ++jd)
        acc[jd] = wmma_bf16f32(ap, b_frag_bf16(Vl, 64, jd * 16, kk * 32, lane), acc[jd]);
    }
    buf ^= 1;
  }

  // ---- normalize, write Y [B,T,C] bf16 ----
#pragma unroll
  for (int r = 0; r < 8; ++r) {
    const int t = q0 + rbase + r;
    const float inv = 1.0f / lrow[r];
#pragma unroll
    for (int jd = 0; jd < 4; ++jd) {
      const int d = jd * 16 + nl;
      Y[(((size_t)b * TT) + t) * CC + h * DD + d] = (__bf16)(acc[jd][r] * inv);
    }
  }
}

// ================= Kernel 3: output projection -> f32 =================
__global__ __launch_bounds__(256) void proj_gemm_kernel(
    const __bf16* __restrict__ Yb, const __bf16* __restrict__ Wpb,
    const float* __restrict__ bproj, float* __restrict__ out)
{
  __shared__ __bf16 As[2][128][64];
  __shared__ __bf16 Bs[2][128][64];

  const int tid  = threadIdx.x;
  const int lane = tid & 31;
  const int wave = tid >> 5;
  const int n0 = blockIdx.x * 128;
  const int m0 = blockIdx.y * 128;
  const int wm = (wave >> 1) * 32;
  const int wn = (wave & 1) * 64;

  const int crow = tid >> 1;
  const int cseg = (tid & 1) * 32;

  auto issue_tile = [&](int k0, int buf) {
    const __bf16* gA = Yb + (size_t)(m0 + crow) * CC + k0 + cseg;
    uint32_t la = lds_off(&As[buf][crow][cseg]);
    async_b128(la,      gA);
    async_b128(la + 16, gA + 8);
    async_b128(la + 32, gA + 16);
    async_b128(la + 48, gA + 24);
    const __bf16* gB = Wpb + (size_t)(n0 + crow) * CC + k0 + cseg;
    uint32_t lb = lds_off(&Bs[buf][crow][cseg]);
    async_b128(lb,      gB);
    async_b128(lb + 16, gB + 8);
    async_b128(lb + 32, gB + 16);
    async_b128(lb + 48, gB + 24);
  };

  v8f acc[2][4];
#pragma unroll
  for (int a = 0; a < 2; ++a)
#pragma unroll
    for (int j = 0; j < 4; ++j) acc[a][j] = v8f_zero();

  int buf = 0;
  issue_tile(0, buf);
  for (int kt = 0; kt < CC / 64; ++kt) {
    wait_async();
    __syncthreads();
    if (kt + 1 < CC / 64) issue_tile((kt + 1) * 64, buf ^ 1);
    const __bf16* Al = &As[buf][0][0];
    const __bf16* Bl = &Bs[buf][0][0];
#pragma unroll
    for (int kk = 0; kk < 2; ++kk) {
      v16bf a0 = a_frag_bf16(Al, 64, wm,      kk * 32, lane);
      v16bf a1 = a_frag_bf16(Al, 64, wm + 16, kk * 32, lane);
#pragma unroll
      for (int j = 0; j < 4; ++j) {
        v16bf b = b_frag_bf16(Bl, 64, wn + 16 * j, kk * 32, lane);
        acc[0][j] = wmma_bf16f32(a0, b, acc[0][j]);
        acc[1][j] = wmma_bf16f32(a1, b, acc[1][j]);
      }
    }
    buf ^= 1;
  }

  const int rbase = (lane >> 4) << 3;
  const int nl = lane & 15;
#pragma unroll
  for (int aT = 0; aT < 2; ++aT) {
#pragma unroll
    for (int j = 0; j < 4; ++j) {
      const int ncol = n0 + wn + 16 * j + nl;
      const float bv = bproj[ncol];
#pragma unroll
      for (int r = 0; r < 8; ++r) {
        const int row = m0 + wm + 16 * aT + rbase + r;
        out[(size_t)row * CC + ncol] = acc[aT][j][r] + bv;
      }
    }
  }
}

extern "C" void kernel_launch(void* const* d_in, const int* in_sizes, int n_in,
                              void* d_out, int out_size, void* d_ws, size_t ws_size,
                              hipStream_t stream) {
  const float* x     = (const float*)d_in[0];
  const float* xt    = (const float*)d_in[1];
  const float* Wattn = (const float*)d_in[2];
  const float* battn = (const float*)d_in[3];
  const float* Wproj = (const float*)d_in[4];
  const float* bproj = (const float*)d_in[5];
  float* out = (float*)d_out;

  const size_t nTok  = (size_t)BB * TT * CC;       // 8.39M (x, Y, Q, K, V each)
  const size_t nWa   = (size_t)3 * CC * CC;        // 3.15M
  const size_t nWp   = (size_t)CC * CC;            // 1.05M

  __bf16* Qh  = (__bf16*)d_ws;
  __bf16* Kh  = Qh  + nTok;
  __bf16* Vh  = Kh  + nTok;   // d-major [B,H,D,T]
  __bf16* Yb  = Vh  + nTok;
  __bf16* xbf = Yb  + nTok;
  __bf16* Wab = xbf + nTok;
  __bf16* Wpb = Wab + nWa;

  dim3 blk(256);
  // one-time f32 -> bf16 conversions (each thread handles 8 elements)
  cvt_bf16_kernel<<<dim3(nTok / 2048), blk, 0, stream>>>(x, xbf);
  cvt_bf16_kernel<<<dim3(nWa  / 2048), blk, 0, stream>>>(Wattn, Wab);
  cvt_bf16_kernel<<<dim3(nWp  / 2048), blk, 0, stream>>>(Wproj, Wpb);

  qkv_gemm_kernel<<<dim3((3 * CC) / 128, (BB * TT) / 128), blk, 0, stream>>>(
      xbf, xt, Wab, battn, Qh, Kh, Vh);
  flash_attn_kernel<<<dim3(TT / 128, BB * HH), blk, 0, stream>>>(Qh, Kh, Vh, Yb);
  proj_gemm_kernel<<<dim3(CC / 128, (BB * TT) / 128), blk, 0, stream>>>(
      Yb, Wpb, bproj, out);
}
